// BigramHashEmbedding_15126874817111
// MI455X (gfx1250) — matrix operand order repaired and checked by reading, not compiled, not run on hardware
//
#include <hip/hip_runtime.h>
#include <hip/hip_bf16.h>

typedef __attribute__((ext_vector_type(16))) __bf16 v16bf;
typedef __attribute__((ext_vector_type(8)))  float  v8f;

#define HIDDEN     2048
#define MODEL_DIM  6144
#define SEQ        4096
#define NTOK       16384
#define VOCAB_MASK 131071u
#define HASH_MULT  1000003u

#define BM 128
#define BN 128
#define BK 32
#define LDSTR 40                 // halves per LDS row: 32 + 8 pad (80B = 20 banks)
#define KSTEPS (HIDDEN / BK)     // 64

#define A_WS_ELEMS ((size_t)NTOK * HIDDEN)        // gathered bf16 rows
#define B_WS_ELEMS ((size_t)MODEL_DIM * HIDDEN)   // bf16 proj
#define WS_NEEDED  ((A_WS_ELEMS + B_WS_ELEMS) * sizeof(unsigned short))

#define HAS_ASYNC_LDS __has_builtin(__builtin_amdgcn_global_load_async_to_lds_b128)

// Pack two fp32 -> two bf16 (RNE) using only 32-bit ALU ops (no b16 subreg moves).
__device__ __forceinline__ unsigned pk2bf(float x, float y) {
  unsigned ux = __float_as_uint(x), uy = __float_as_uint(y);
  ux += 0x7FFFu + ((ux >> 16) & 1u);
  uy += 0x7FFFu + ((uy >> 16) & 1u);
  return (ux >> 16) | (uy & 0xFFFF0000u);
}

#if HAS_ASYNC_LDS
// Builtin prototype (from hipcc diagnostic): (v4i addrspace(1)*, v4i addrspace(3)*, Imm, Imm)
typedef int v4i_vs __attribute__((vector_size(16)));
typedef __attribute__((address_space(1))) v4i_vs* gptr_b128;
typedef __attribute__((address_space(3))) v4i_vs* lptr_b128;

// 16B async copy global -> LDS (ASYNCcnt-tracked, no VGPR transit).
// Low 32 bits of a generic LDS pointer are the LDS offset (aperture rules),
// so the uintptr round-trip recovers the addrspace(3) value.
__device__ __forceinline__ void async_cp16(const unsigned short* g, unsigned short* l) {
  __builtin_amdgcn_global_load_async_to_lds_b128(
      (gptr_b128)(uintptr_t)g,
      (lptr_b128)(unsigned)(uintptr_t)l,
      0, 0);
}
__device__ __forceinline__ void wait_async0() {
#if __has_builtin(__builtin_amdgcn_s_wait_asynccnt)
  __builtin_amdgcn_s_wait_asynccnt(0);
#else
  asm volatile("s_wait_asynccnt 0x0" ::: "memory");
#endif
}
#endif

// ---------------------------------------------------------------------------
// Phase 1: hash + gather + convert-to-bf16, done ONCE per row (not per tile).
// blocks [0, NTOK): A rows (gathered from table); [NTOK, NTOK+MODEL_DIM): B rows
// ---------------------------------------------------------------------------
__global__ __launch_bounds__(256)
void prep_convert(const int* __restrict__ ids, const float* __restrict__ table,
                  const float* __restrict__ proj, unsigned short* __restrict__ ws) {
  const int row = blockIdx.x;
  const int t = threadIdx.x;
  const float* src;
  unsigned short* dst;
  if (row < NTOK) {
    unsigned cur  = (unsigned)ids[row];
    unsigned prev = ((row & (SEQ - 1)) == 0) ? 0u : (unsigned)ids[row - 1];
    unsigned idx  = (prev * HASH_MULT + cur) & VOCAB_MASK;  // exact: mod 2^17
    src = table + (size_t)idx * HIDDEN;
    dst = ws + (size_t)row * HIDDEN;
  } else {
    int r = row - NTOK;
    src = proj + (size_t)r * HIDDEN;
    dst = ws + A_WS_ELEMS + (size_t)r * HIDDEN;
  }
  // 2048 floats / 256 threads = 8 floats -> one 16B bf16 store per thread
  const float4 f0 = *(const float4*)(src + t * 8);
  const float4 f1 = *(const float4*)(src + t * 8 + 4);
  uint4 o;
  o.x = pk2bf(f0.x, f0.y); o.y = pk2bf(f0.z, f0.w);
  o.z = pk2bf(f1.x, f1.y); o.w = pk2bf(f1.z, f1.w);
  *(uint4*)(dst + t * 8) = o;
}

// ---------------------------------------------------------------------------
// Shared WMMA tile compute: 8x v_wmma_f32_16x16x32_bf16 per wave per K-step
// ---------------------------------------------------------------------------
__device__ __forceinline__ void tile_compute(const unsigned short* __restrict__ A,
                                             int wm, int wn, int laneLo, int laneHi,
                                             v8f acc[4][2]) {
  const unsigned short* B = A + BM * LDSTR;
  v16bf af[4], bfr[2];
  // A 16x32 bf16: lanes 0-15 hold K 0-7 & 16-23, lanes 16-31 hold K 8-15 & 24-31
#pragma unroll
  for (int mt = 0; mt < 4; ++mt) {
    const unsigned short* p = A + (wm + mt * 16 + laneLo) * LDSTR + laneHi * 8;
    uint4* d = (uint4*)&af[mt];
    d[0] = *(const uint4*)p;
    d[1] = *(const uint4*)(p + 16);
  }
  // B 32x16 bf16: lanes 0-15 hold K 0-7 & 16-23, lanes 16-31 hold K 8-15 & 24-31 at column lane%16
#pragma unroll
  for (int nt = 0; nt < 2; ++nt) {
    const unsigned short* p = B + (wn + nt * 16 + laneLo) * LDSTR + laneHi * 8;
    uint4* d = (uint4*)&bfr[nt];
    d[0] = *(const uint4*)p;
    d[1] = *(const uint4*)(p + 16);
  }
#pragma unroll
  for (int mt = 0; mt < 4; ++mt)
#pragma unroll
    for (int nt = 0; nt < 2; ++nt)
      acc[mt][nt] = __builtin_amdgcn_wmma_f32_16x16x32_bf16(
          false, af[mt], false, bfr[nt], (short)0, acc[mt][nt], false, false);
}

__device__ __forceinline__ void write_out(float* __restrict__ out, int m0, int n0,
                                          int wm, int wn, int laneLo, int laneHi,
                                          v8f acc[4][2]) {
  float* o = out + (size_t)m0 * MODEL_DIM + n0;
#pragma unroll
  for (int mt = 0; mt < 4; ++mt)
#pragma unroll
    for (int nt = 0; nt < 2; ++nt)
#pragma unroll
      for (int v = 0; v < 8; ++v) {
        int row = wm + mt * 16 + laneHi * 8 + v;   // C/D: M = v + 8*(lane>=16)
        int col = wn + nt * 16 + laneLo;           //      N = lane%16
        o[(size_t)row * MODEL_DIM + col] = acc[mt][nt][v];
      }
}

// ---------------------------------------------------------------------------
// Phase 2: pure-bf16 GEMM  out[16384 x 6144] = A[16384 x 2048] * B[6144 x 2048]^T
// 256 threads = 8 waves; block tile 128x128; K-step 32; double-buffered LDS,
// staged via GLOBAL_LOAD_ASYNC_TO_LDS_B128 when available.
// ---------------------------------------------------------------------------
__global__ __launch_bounds__(256)
void gemm_bf16(const unsigned short* __restrict__ Abf,
               const unsigned short* __restrict__ Bbf,
               float* __restrict__ out) {
  __shared__ __align__(16) unsigned short lmem[2 * 2 * BM * LDSTR];

  const int t  = threadIdx.x;
  const int m0 = blockIdx.y * BM;
  const int n0 = blockIdx.x * BN;

  // staging: thread t owns (rows rb, rb+64; 16B chunk kc4 of the 64B row segment)
  const int kc4 = t & 3;
  const int rb  = t >> 2;        // 0..63
  const unsigned short* arow0 = Abf + (size_t)(m0 + rb) * HIDDEN + kc4 * 8;
  const unsigned short* arow1 = arow0 + (size_t)64 * HIDDEN;
  const unsigned short* brow0 = Bbf + (size_t)(n0 + rb) * HIDDEN + kc4 * 8;
  const unsigned short* brow1 = brow0 + (size_t)64 * HIDDEN;

  const int lane = t & 31, wid = t >> 5;
  const int laneLo = lane & 15, laneHi = lane >> 4;
  const int wm = (wid >> 2) * 64;
  const int wn = (wid & 3) * 32;

  v8f acc[4][2];
#pragma unroll
  for (int mt = 0; mt < 4; ++mt)
#pragma unroll
    for (int nt = 0; nt < 2; ++nt)
#pragma unroll
      for (int v = 0; v < 8; ++v) acc[mt][nt][v] = 0.f;

#if HAS_ASYNC_LDS
  unsigned short* la0 = lmem + rb * LDSTR + kc4 * 8;
  unsigned short* la1 = lmem + (rb + 64) * LDSTR + kc4 * 8;
  unsigned short* lb0 = la0 + BM * LDSTR;
  unsigned short* lb1 = la1 + BM * LDSTR;
  auto issue_async = [&](int buf, int k0) {
    const int o = buf * (2 * BM * LDSTR);
    async_cp16(arow0 + k0, la0 + o);
    async_cp16(arow1 + k0, la1 + o);
    async_cp16(brow0 + k0, lb0 + o);
    async_cp16(brow1 + k0, lb1 + o);
  };

  issue_async(0, 0);
  wait_async0();
  __syncthreads();
#pragma unroll 2
  for (int ks = 0; ks < KSTEPS - 1; ++ks) {
    issue_async((ks + 1) & 1, (ks + 1) * BK);   // DMA idle buffer, overlaps WMMA
    tile_compute(lmem + (ks & 1) * (2 * BM * LDSTR), wm, wn, laneLo, laneHi, acc);
    wait_async0();                              // own copies landed in LDS
    __syncthreads();                            // all waves' copies visible
  }
  tile_compute(lmem + ((KSTEPS - 1) & 1) * (2 * BM * LDSTR), wm, wn, laneLo, laneHi, acc);
#else
  uint4 sa0, sa1, sb0, sb1;
  auto stage_load = [&](int k0) {
    sa0 = *(const uint4*)(arow0 + k0);
    sa1 = *(const uint4*)(arow1 + k0);
    sb0 = *(const uint4*)(brow0 + k0);
    sb1 = *(const uint4*)(brow1 + k0);
  };
  auto stage_store = [&](int buf) {
    unsigned short* A = lmem + buf * (2 * BM * LDSTR);
    unsigned short* B = A + BM * LDSTR;
    *(uint4*)(A + rb * LDSTR + kc4 * 8)        = sa0;
    *(uint4*)(A + (rb + 64) * LDSTR + kc4 * 8) = sa1;
    *(uint4*)(B + rb * LDSTR + kc4 * 8)        = sb0;
    *(uint4*)(B + (rb + 64) * LDSTR + kc4 * 8) = sb1;
  };
  stage_load(0);
  stage_store(0);
  __syncthreads();
#pragma unroll 2
  for (int ks = 0; ks < KSTEPS - 1; ++ks) {
    stage_load((ks + 1) * BK);
    tile_compute(lmem + (ks & 1) * (2 * BM * LDSTR), wm, wn, laneLo, laneHi, acc);
    stage_store((ks + 1) & 1);
    __syncthreads();
  }
  tile_compute(lmem + ((KSTEPS - 1) & 1) * (2 * BM * LDSTR), wm, wn, laneLo, laneHi, acc);
#endif

  write_out(out, m0, n0, wm, wn, laneLo, laneHi, acc);
}

// ---------------------------------------------------------------------------
// Fallback (ws too small): fused gather+convert+GEMM, conversions via pk2bf.
// ---------------------------------------------------------------------------
__global__ __launch_bounds__(256)
void bigram_fused(const int* __restrict__ ids, const float* __restrict__ table,
                  const float* __restrict__ proj, float* __restrict__ out) {
  __shared__ __align__(16) unsigned short lmem[2 * 2 * BM * LDSTR];
  __shared__ unsigned sidx[BM];

  const int t  = threadIdx.x;
  const int m0 = blockIdx.y * BM;
  const int n0 = blockIdx.x * BN;

  if (t < BM) {
    int m = m0 + t;
    unsigned cur  = (unsigned)ids[m];
    unsigned prev = ((m & (SEQ - 1)) == 0) ? 0u : (unsigned)ids[m - 1];
    sidx[t] = (prev * HASH_MULT + cur) & VOCAB_MASK;
  }
  __syncthreads();

  const int kc = t & 7;
  const int rbase = t >> 3;
  const float* arow[4];
  const float* brow[4];
#pragma unroll
  for (int i = 0; i < 4; ++i) {
    int r = rbase + i * 32;
    arow[i] = table + (size_t)sidx[r] * HIDDEN + kc * 4;
    brow[i] = proj + (size_t)(n0 + r) * HIDDEN + kc * 4;
  }

  const int lane = t & 31, wid = t >> 5;
  const int laneLo = lane & 15, laneHi = lane >> 4;
  const int wm = (wid >> 2) * 64;
  const int wn = (wid & 3) * 32;

  v8f acc[4][2];
#pragma unroll
  for (int mt = 0; mt < 4; ++mt)
#pragma unroll
    for (int nt = 0; nt < 2; ++nt)
#pragma unroll
      for (int v = 0; v < 8; ++v) acc[mt][nt][v] = 0.f;

  float4 areg[4], breg[4];
  auto stage_load = [&](int k0) {
#pragma unroll
    for (int i = 0; i < 4; ++i) {
      areg[i] = *(const float4*)(arow[i] + k0);
      breg[i] = *(const float4*)(brow[i] + k0);
    }
  };
  auto stage_store = [&](int buf) {
    unsigned short* A = lmem + buf * (2 * BM * LDSTR);
    unsigned short* B = A + BM * LDSTR;
#pragma unroll
    for (int i = 0; i < 4; ++i) {
      int r = rbase + i * 32;
      *(uint2*)(A + r * LDSTR + kc * 4) =
          make_uint2(pk2bf(areg[i].x, areg[i].y), pk2bf(areg[i].z, areg[i].w));
      *(uint2*)(B + r * LDSTR + kc * 4) =
          make_uint2(pk2bf(breg[i].x, breg[i].y), pk2bf(breg[i].z, breg[i].w));
    }
  };

  stage_load(0);
  stage_store(0);
  __syncthreads();
#pragma unroll 2
  for (int ks = 0; ks < KSTEPS - 1; ++ks) {
    stage_load((ks + 1) * BK);
    tile_compute(lmem + (ks & 1) * (2 * BM * LDSTR), wm, wn, laneLo, laneHi, acc);
    stage_store((ks + 1) & 1);
    __syncthreads();
  }
  tile_compute(lmem + ((KSTEPS - 1) & 1) * (2 * BM * LDSTR), wm, wn, laneLo, laneHi, acc);

  write_out(out, m0, n0, wm, wn, laneLo, laneHi, acc);
}

extern "C" void kernel_launch(void* const* d_in, const int* in_sizes, int n_in,
                              void* d_out, int out_size, void* d_ws, size_t ws_size,
                              hipStream_t stream) {
  (void)in_sizes; (void)n_in; (void)out_size;
  const int*   ids   = (const int*)d_in[0];
  const float* table = (const float*)d_in[1];
  const float* proj  = (const float*)d_in[2];
  float*       out   = (float*)d_out;
  dim3 grid(MODEL_DIM / BN, NTOK / BM);   // (48, 128)

  if (ws_size >= WS_NEEDED) {
    unsigned short* ws = (unsigned short*)d_ws;
    prep_convert<<<NTOK + MODEL_DIM, 256, 0, stream>>>(ids, table, proj, ws);
    gemm_bf16<<<grid, 256, 0, stream>>>(ws, ws + A_WS_ELEMS, out);
  } else {
    bigram_fused<<<grid, 256, 0, stream>>>(ids, table, proj, out);
  }
}
